// TransAttLoss_15702400434327
// MI455X (gfx1250) — compile-verified, matrix operand order checked by reading, not run on hardware
//
#include <hip/hip_runtime.h>
#include <hip/hip_bf16.h>

typedef __attribute__((ext_vector_type(16))) _Float16 v16h;
typedef __attribute__((ext_vector_type(8)))  float    v8f;
typedef __attribute__((ext_vector_type(4)))  float    f32x4;
typedef __attribute__((ext_vector_type(4)))  int      i32x4;

#define HW_   262144                  // 512*512
#define NPIX_ 8388608LL               // 32*512*512
#define NGRP_ 2097152LL               // NPIX/4 (4-pixel groups)
#define SMOOTH_ 1e-6

// ws layout (13 doubles): [0]=ce_sum, [1..4]=intersection[c], [5..8]=probsum[c], [9..12]=count[c]
__global__ void tal_zero_ws(double* __restrict__ ws) {
  if (threadIdx.x < 13) ws[threadIdx.x] = 0.0;
}

__global__ __launch_bounds__(256)
void tal_partial(const float* __restrict__ logits,
                 const int*   __restrict__ tgt,
                 double*      __restrict__ ws) {
  float ce = 0.f;
  float it0 = 0.f, it1 = 0.f, it2 = 0.f, it3 = 0.f;   // intersection (= p_t for class t)
  float ps0 = 0.f, ps1 = 0.f, ps2 = 0.f, ps3 = 0.f;   // prob sums per class
  float cn0 = 0.f, cn1 = 0.f, cn2 = 0.f, cn3 = 0.f;   // one-hot counts per class

  const long long stride = (long long)gridDim.x * blockDim.x;
  for (long long g = (long long)blockIdx.x * blockDim.x + threadIdx.x; g < NGRP_; g += stride) {
    int b  = (int)(g >> 16);                 // HW/4 = 65536 groups per batch image
    int hw = ((int)g & 65535) << 2;          // 16B-aligned pixel offset
    size_t base = ((size_t)b * 4) * (size_t)HW_ + (size_t)hw;
    f32x4 x0 = __builtin_nontemporal_load((const f32x4*)(logits + base));
    f32x4 x1 = __builtin_nontemporal_load((const f32x4*)(logits + base + (size_t)HW_));
    f32x4 x2 = __builtin_nontemporal_load((const f32x4*)(logits + base + 2*(size_t)HW_));
    f32x4 x3 = __builtin_nontemporal_load((const f32x4*)(logits + base + 3*(size_t)HW_));
    i32x4 t4 = __builtin_nontemporal_load((const i32x4*)(tgt + (size_t)g * 4));
#pragma unroll
    for (int j = 0; j < 4; ++j) {
      float a0 = x0[j], a1 = x1[j], a2 = x2[j], a3 = x3[j];
      int   t  = t4[j];
      float m  = fmaxf(fmaxf(a0, a1), fmaxf(a2, a3));
      float e0 = __expf(a0 - m), e1 = __expf(a1 - m);
      float e2 = __expf(a2 - m), e3 = __expf(a3 - m);
      float s   = (e0 + e1) + (e2 + e3);
      float inv = __builtin_amdgcn_rcpf(s);
      float p0 = e0 * inv, p1 = e1 * inv, p2 = e2 * inv, p3 = e3 * inv;
      float at = (t == 0) ? a0 : ((t == 1) ? a1 : ((t == 2) ? a2 : a3));
      ce += (m + __logf(s)) - at;            // -log_softmax at target class
      ps0 += p0; ps1 += p1; ps2 += p2; ps3 += p3;
      if      (t == 0) { it0 += p0; cn0 += 1.f; }
      else if (t == 1) { it1 += p1; cn1 += 1.f; }
      else if (t == 2) { it2 += p2; cn2 += 1.f; }
      else             { it3 += p3; cn3 += 1.f; }
    }
  }

  // Stage 13 per-thread partials (padded to 16 slots) into LDS: smem[thread][slot]
  __shared__ float smem[256 * 16];
  {
    float* row = &smem[threadIdx.x * 16];
    row[0]  = ce;
    row[1]  = it0; row[2]  = it1; row[3]  = it2; row[4]  = it3;
    row[5]  = ps0; row[6]  = ps1; row[7]  = ps2; row[8]  = ps3;
    row[9]  = cn0; row[10] = cn1; row[11] = cn2; row[12] = cn3;
    row[13] = 0.f; row[14] = 0.f; row[15] = 0.f;
  }
  __syncthreads();

  // Wave 0 reduces the 256x16 partial matrix with WMMA:
  // A = ones(16x32 f16), B = 32 threads x 16 slots  =>  D[m][n] = sum_k B[k][n].
  // Exact-ish f32 via hi/lo f16 split (two WMMAs per chunk, accumulated in f32 C).
  if (threadIdx.x < 32) {
    v16h ones;
#pragma unroll
    for (int i = 0; i < 16; ++i) ones[i] = (_Float16)1.0f;
    v8f acc = {};
    const int L      = threadIdx.x;
    const int slot   = L & 15;        // column n this lane feeds
    const int rowoff = L & 16;        // lanes 16-31 supply K=16..31
#pragma unroll
    for (int q = 0; q < 8; ++q) {     // 8 chunks of 32 threads
      v16h bh, bl;
      const int rowbase = q * 32 + rowoff;
#pragma unroll
      for (int e = 0; e < 16; ++e) {
        float x = smem[(rowbase + e) * 16 + slot];
        _Float16 h = (_Float16)x;
        bh[e] = h;
        bl[e] = (_Float16)(x - (float)h);
      }
      acc = __builtin_amdgcn_wmma_f32_16x16x32_f16(false, ones, false, bh,
                                                   (short)0, acc, false, false);
      acc = __builtin_amdgcn_wmma_f32_16x16x32_f16(false, ones, false, bl,
                                                   (short)0, acc, false, false);
    }
    // All rows of D are identical; lane n, VGPR0 = block total for slot n.
    if (L < 13) atomicAdd(&ws[L], (double)acc[0]);
  }
}

__global__ void tal_finalize(const double* __restrict__ ws, float* __restrict__ out) {
  if (threadIdx.x == 0 && blockIdx.x == 0) {
    double ce_loss = ws[0] / (double)NPIX_;
    double dsum = 0.0;
#pragma unroll
    for (int c = 0; c < 4; ++c) {
      double I = ws[1 + c];
      double U = ws[5 + c] + ws[9 + c];
      dsum += 1.0 - (2.0 * I + SMOOTH_) / (U + SMOOTH_);
    }
    double dice_loss = dsum * 0.25;
    out[0] = (float)(0.5 * ce_loss + 0.5 * dice_loss);
    out[1] = (float)(1.0 - dice_loss);
  }
}

extern "C" void kernel_launch(void* const* d_in, const int* in_sizes, int n_in,
                              void* d_out, int out_size, void* d_ws, size_t ws_size,
                              hipStream_t stream) {
  const float* logits = (const float*)d_in[0];
  const int*   tgt    = (const int*)d_in[1];
  double*      ws     = (double*)d_ws;
  float*       out    = (float*)d_out;

  tal_zero_ws<<<1, 32, 0, stream>>>(ws);
  tal_partial<<<2048, 256, 0, stream>>>(logits, tgt, ws);
  tal_finalize<<<1, 1, 0, stream>>>(ws, out);
}
